// RandomProjectionQuantizer_26611617366415
// MI455X (gfx1250) — matrix operand order repaired and checked by reading, not compiled, not run on hardware
//
#include <hip/hip_runtime.h>
#include <stdint.h>

// ---- problem constants (from reference setup_inputs) ----
#define B_ 4
#define N_ 4096
#define D_ 1024
#define H_ 4
#define E_ 256
#define C_ 2048
#define M_ (B_ * N_)          // 16384 rows total
#define LDS_PAD 8
#define LDA (D_ + LDS_PAD)    // 1032 bf16/row -> LDS row stride 2064B (conflict-free)
#define CC 128                // codebook rows per LDS chunk in sim kernel
#define LDB (E_ + 8)          // 264 bf16/row -> LDS row stride 528B (conflict-free)

typedef __attribute__((ext_vector_type(16))) __bf16 v16bf;
typedef __attribute__((ext_vector_type(8)))  __bf16 v8bf;
typedef __attribute__((ext_vector_type(8)))  float  v8f;

union AFrag { v16bf v; v8bf half[2]; };

// f32 -> bf16 (round to nearest even), bit-level
__device__ __forceinline__ unsigned short f2bf_bits(float f) {
  union { float f; uint32_t u; } c; c.f = f;
  uint32_t u = c.u;
  u += 0x7FFFu + ((u >> 16) & 1u);
  return (unsigned short)(u >> 16);
}

// ---------------------------------------------------------------------------
// Kernel 1: rand_projs [h][d][e] f32  ->  Pt [h][e][d] bf16 (K-contiguous B)
// ---------------------------------------------------------------------------
__global__ void k_prep_proj(const float* __restrict__ P,
                            unsigned short* __restrict__ Pt) {
  int idx = blockIdx.x * blockDim.x + threadIdx.x;   // over H*D*E
  int h = idx / (D_ * E_);
  int r = idx - h * (D_ * E_);
  int d = r / E_;
  int e = r - d * E_;
  Pt[((size_t)(h * E_ + e)) * D_ + d] = f2bf_bits(P[idx]);
}

// ---------------------------------------------------------------------------
// Kernel 2: codebook [h][c][e] f32 -> L2-normalized bf16 rows. 1 wave per row.
// ---------------------------------------------------------------------------
__global__ void k_prep_cb(const float* __restrict__ CB,
                          unsigned short* __restrict__ CBn) {
  int row  = (blockIdx.x * blockDim.x + threadIdx.x) >> 5;  // over H_*C_
  int lane = threadIdx.x & 31;
  const float* r = CB + (size_t)row * E_;
  float vals[8];
  float ss = 0.f;
#pragma unroll
  for (int i = 0; i < 8; ++i) {
    float v = r[lane + i * 32];
    vals[i] = v;
    ss += v * v;
  }
#pragma unroll
  for (int m = 16; m >= 1; m >>= 1) ss += __shfl_xor(ss, m, 32);
  float scale = 1.0f / fmaxf(sqrtf(ss), 1e-12f);
  unsigned short* o = CBn + (size_t)row * E_;
#pragma unroll
  for (int i = 0; i < 8; ++i) o[lane + i * 32] = f2bf_bits(vals[i] * scale);
}

// ---------------------------------------------------------------------------
// Kernel 3: proj GEMM.  Block = 128 thr (4 waves = 4 heads), tile = 32 rows.
// x tile staged once in LDS as bf16; each B fragment feeds 2 row-tiles.
// ks loop kept rolled (unroll 2) so A fragments stream from LDS instead of
// being hoisted into ~512 VGPRs (which would drop occupancy to 1 wave/SIMD).
// ---------------------------------------------------------------------------
__global__ void __launch_bounds__(128) k_proj_gemm(
    const float* __restrict__ x, const unsigned short* __restrict__ Pt,
    unsigned short* __restrict__ proj) {
  __shared__ __align__(16) unsigned short sA[32 * LDA];   // 66 KB
  const int tid = threadIdx.x;
  const int m0  = blockIdx.x * 32;

  // cooperative load+convert of 32x1024 f32 tile -> bf16 LDS
  const float4* x4 = (const float4*)x;
#pragma unroll 4
  for (int i = 0; i < 64; ++i) {
    int f4 = i * 128 + tid;         // 0..8191 float4 slots
    int r  = f4 >> 8;               // 256 float4 per row
    int c4 = f4 & 255;
    float4 v = x4[(size_t)(m0 + r) * (D_ / 4) + c4];
    union { unsigned short s[4]; uint64_t u; } pk;
    pk.s[0] = f2bf_bits(v.x); pk.s[1] = f2bf_bits(v.y);
    pk.s[2] = f2bf_bits(v.z); pk.s[3] = f2bf_bits(v.w);
    *(uint64_t*)&sA[r * LDA + c4 * 4] = pk.u;
  }
  __syncthreads();

  const int head = tid >> 5;           // wave id == head
  const int lane = tid & 31;
  const int arow = lane & 15;          // A: lane -> M row (within tile)
  const int asel = (lane >> 4) * 8;    // A: K sub-offset {0,8} (+16 for upper)
  const int bcol = lane & 15;          // B: lane -> N column
  const int bsel = (lane >> 4) * 8;    // B: K sub-offset {0,8} (+16 for upper)

#pragma unroll 1
  for (int eg = 0; eg < 4; ++eg) {     // 4 column groups of 64
    const int ebase = eg * 64;
    v8f acc[2][4] = {};
#pragma unroll 2
    for (int ks = 0; ks < 32; ++ks) {
      const int k0 = ks * 32;
      AFrag a[2];
#pragma unroll
      for (int rt = 0; rt < 2; ++rt) {
        const unsigned short* ap = &sA[(rt * 16 + arow) * LDA + k0 + asel];
        a[rt].half[0] = *(const v8bf*)(ap);        // ds_load_b128
        a[rt].half[1] = *(const v8bf*)(ap + 16);
      }
#pragma unroll
      for (int t = 0; t < 4; ++t) {
        const int e = ebase + t * 16 + bcol;
        const unsigned short* bp =
            Pt + ((size_t)(head * E_ + e)) * D_ + k0 + bsel;
        AFrag b;
        b.half[0] = *(const v8bf*)(bp);            // global_load_b128 (L2)
        b.half[1] = *(const v8bf*)(bp + 16);
        acc[0][t] = __builtin_amdgcn_wmma_f32_16x16x32_bf16(
            false, a[0].v, false, b.v, (short)0, acc[0][t], false, false);
        acc[1][t] = __builtin_amdgcn_wmma_f32_16x16x32_bf16(
            false, a[1].v, false, b.v, (short)0, acc[1][t], false, false);
      }
    }
    // D layout: VGPR v, lane -> (M = v + 8*(lane>>4), N = lane&15)
#pragma unroll
    for (int rt = 0; rt < 2; ++rt) {
      const int mrow = m0 + rt * 16 + ((lane >> 4) * 8);
#pragma unroll
      for (int t = 0; t < 4; ++t) {
        const int e = ebase + t * 16 + bcol;
#pragma unroll
        for (int v = 0; v < 8; ++v) {
          proj[((size_t)(head * M_ + (mrow + v))) * E_ + e] =
              f2bf_bits(acc[rt][t][v]);
        }
      }
    }
  }
}

// ---------------------------------------------------------------------------
// Kernel 4: sim GEMM + argmax.  Block = 256 thr (8 waves, same head,
// 8 consecutive 16-row tiles).  Codebook staged through LDS in 128-row
// chunks so each chunk is read from L2 once per 128 rows of x.
// ---------------------------------------------------------------------------
__global__ void __launch_bounds__(256) k_sim_argmax(
    const unsigned short* __restrict__ proj,
    const unsigned short* __restrict__ CBn,
    int* __restrict__ out) {
  __shared__ __align__(16) unsigned short sB[CC * LDB];   // ~67 KB
  const int tid  = threadIdx.x;
  const int head = blockIdx.y;
  const int wv   = tid >> 5;
  const int lane = tid & 31;
  const int m0   = blockIdx.x * 128 + wv * 16;
  const int arow = lane & 15;
  const int asel = (lane >> 4) * 8;
  const int bcol = lane & 15;
  const int bsel = (lane >> 4) * 8;

  // preload this wave's A fragments (16x256 proj tile, 8 k-steps)
  AFrag A[8];
  const unsigned short* aprow =
      proj + ((size_t)(head * M_ + (m0 + arow))) * E_;
#pragma unroll
  for (int ks = 0; ks < 8; ++ks) {
    const unsigned short* ap = aprow + ks * 32 + asel;
    A[ks].half[0] = *(const v8bf*)(ap);
    A[ks].half[1] = *(const v8bf*)(ap + 16);
  }

  float bestv[8];
  int   besti[8];
#pragma unroll
  for (int v = 0; v < 8; ++v) { bestv[v] = -3.4e38f; besti[v] = 0; }

  const unsigned short* cbh = CBn + (size_t)head * C_ * E_;
#pragma unroll 1
  for (int ch = 0; ch < C_ / CC; ++ch) {        // 16 chunks of 128 codes
    // cooperative stage: 128 rows x 256 bf16 = 64KB -> padded LDS
    const unsigned short* src = cbh + (size_t)ch * CC * E_;
#pragma unroll 4
    for (int i = 0; i < 16; ++i) {
      int idx = i * 256 + tid;                  // 0..4095 b128 slots
      int r   = idx >> 5;                       // 32 b128 per row
      int c16 = idx & 31;
      v8bf val = *(const v8bf*)(src + r * E_ + c16 * 8);
      *(v8bf*)&sB[r * LDB + c16 * 8] = val;
    }
    __syncthreads();

#pragma unroll 1
    for (int ct = 0; ct < CC / 16; ++ct) {      // 8 c-tiles per chunk
      v8f acc = {};
      const unsigned short* brow = &sB[(ct * 16 + bcol) * LDB];
#pragma unroll
      for (int ks = 0; ks < 8; ++ks) {
        AFrag b;
        const unsigned short* bp = brow + ks * 32 + bsel;
        b.half[0] = *(const v8bf*)(bp);         // ds_load_b128
        b.half[1] = *(const v8bf*)(bp + 16);
        acc = __builtin_amdgcn_wmma_f32_16x16x32_bf16(
            false, A[ks].v, false, b.v, (short)0, acc, false, false);
      }
      const int cidx = ch * CC + ct * 16 + bcol;
#pragma unroll
      for (int v = 0; v < 8; ++v) {
        float s = acc[v];
        if (s > bestv[v]) { bestv[v] = s; besti[v] = cidx; }  // first-max wins
      }
    }
    __syncthreads();
  }

  // argmax across the 16 lanes holding the 16 columns (each half separately)
#pragma unroll
  for (int mask = 8; mask >= 1; mask >>= 1) {
#pragma unroll
    for (int v = 0; v < 8; ++v) {
      float ov = __shfl_xor(bestv[v], mask, 16);
      int   oi = __shfl_xor(besti[v], mask, 16);
      if (ov > bestv[v] || (ov == bestv[v] && oi < besti[v])) {
        bestv[v] = ov; besti[v] = oi;
      }
    }
  }
  if (bcol == 0) {  // lanes 0 (rows m0..m0+7) and 16 (rows m0+8..m0+15)
    const int mrow = m0 + ((lane >> 4) * 8);
#pragma unroll
    for (int v = 0; v < 8; ++v) out[(mrow + v) * H_ + head] = besti[v];
  }
}

// ---------------------------------------------------------------------------
extern "C" void kernel_launch(void* const* d_in, const int* in_sizes, int n_in,
                              void* d_out, int out_size, void* d_ws,
                              size_t ws_size, hipStream_t stream) {
  const float* x  = (const float*)d_in[0];   // [4,4096,1024]
  const float* P  = (const float*)d_in[1];   // [4,1024,256]
  const float* CB = (const float*)d_in[2];   // [4,2048,256]
  int* out = (int*)d_out;                    // [4,4096,4] int32 indices

  // workspace layout (bytes):
  //   Pt   bf16 [H][E][D]  : 2 MiB   @ 0
  //   CBn  bf16 [H][C][E]  : 4 MiB   @ 2 MiB
  //   proj bf16 [H][M][E]  : 32 MiB  @ 6 MiB
  char* ws = (char*)d_ws;
  unsigned short* Pt   = (unsigned short*)(ws);
  unsigned short* CBn  = (unsigned short*)(ws + (2u << 20));
  unsigned short* proj = (unsigned short*)(ws + (6u << 20));

  k_prep_proj<<<dim3((H_ * D_ * E_) / 256), 256, 0, stream>>>(P, Pt);
  k_prep_cb<<<dim3((H_ * C_) / 8), 256, 0, stream>>>(CB, CBn);
  k_proj_gemm<<<dim3(M_ / 32), 128, 0, stream>>>(x, Pt, proj);
  k_sim_argmax<<<dim3(M_ / 128, H_), 256, 0, stream>>>(proj, CBn, out);
}
